// FlexAttentionCPB_35673998360606
// MI455X (gfx1250) — matrix operand order repaired and checked by reading, not compiled, not run on hardware
//
#include <hip/hip_runtime.h>
#include <hip/hip_bf16.h>
#include <stdint.h>

#define B_   4
#define H_   12
#define N_   1025
#define D_   64
#define HID_ 32
#define TBL_ 3969   // (2*32-1)^2

typedef __attribute__((ext_vector_type(16))) _Float16 v16h;
typedef __attribute__((ext_vector_type(2)))  __fp16   fp16x2;
typedef __attribute__((ext_vector_type(8)))  float    v8f;
typedef __attribute__((ext_vector_type(4)))  float    f32x4;
typedef __attribute__((ext_vector_type(4)))  uint32_t u32x4;
typedef __attribute__((ext_vector_type(2)))  uint32_t u32x2;

__device__ __forceinline__ v8f wmma_f16(v16h a, v16h b, v8f c) {
  // D = A(16x32 f16) * B(32x16 f16) + C(16x16 f32)
  return __builtin_amdgcn_wmma_f32_16x16x32_f16(
      /*neg_a=*/false, a, /*neg_b=*/false, b,
      /*c_mod=*/(short)0, c, /*reuse_a=*/false, /*reuse_b=*/false);
}

// One v_cvt_pk_rtz_f16_f32: pack two f32 into an f16x2 dword.
__device__ __forceinline__ uint32_t pk2(float a, float b) {
  union { fp16x2 h; uint32_t u; } cv;
  cv.h = __builtin_amdgcn_cvt_pkrtz(a, b);
  return cv.u;
}

// VALU cross-half swap: lane L <-> lane L^16 (v_permlanex16_b32, identity sel).
__device__ __forceinline__ uint32_t xs16(uint32_t x) {
  return __builtin_amdgcn_permlanex16(x, x, 0x76543210u, 0xfedcba98u, false, false);
}
__device__ __forceinline__ float xs16f(float x) {
  union { float f; uint32_t u; } a, r;
  a.f = x; r.u = xs16(a.u);
  return r.f;
}

// WMMA A-operand from an LDS f16 row:
// halves 0..7  <- p[dlo .. dlo+7]      (one ds_load_b128)
// halves 8..15 <- p[dlo+16 .. dlo+23]  (one ds_load_b128)
__device__ __forceinline__ v16h ldsA(const _Float16* p, int dlo) {
  u32x4 lo = *(const u32x4*)(p + dlo);
  u32x4 hi = *(const u32x4*)(p + dlo + 16);
  union { v16h v; u32x4 q[2]; } u;
  u.q[0] = lo; u.q[1] = hi;
  return u.v;
}

// ---------------------------------------------------------------------------
// Kernel 1: bt[r][h] = (gelu(rel_table @ w1 + b1) @ w2)[r][h]   (3969 x 12)
// ---------------------------------------------------------------------------
__global__ void cpb_mlp_kernel(const float* __restrict__ rel,
                               const float* __restrict__ w1,
                               const float* __restrict__ b1,
                               const float* __restrict__ w2,
                               float* __restrict__ bt) {
  int r = blockIdx.x * blockDim.x + threadIdx.x;
  if (r >= TBL_) return;
  float r0 = rel[2 * r + 0];
  float r1 = rel[2 * r + 1];
  float acc[H_];
#pragma unroll
  for (int hh = 0; hh < H_; ++hh) acc[hh] = 0.0f;
  for (int j = 0; j < HID_; ++j) {
    float x = fmaf(r0, w1[j], fmaf(r1, w1[HID_ + j], b1[j]));
    float g = 0.5f * x * (1.0f + erff(x * 0.70710678118654752f));  // exact gelu
#pragma unroll
    for (int hh = 0; hh < H_; ++hh) acc[hh] = fmaf(g, w2[j * H_ + hh], acc[hh]);
  }
#pragma unroll
  for (int hh = 0; hh < H_; ++hh) bt[r * H_ + hh] = acc[hh];
}

// ---------------------------------------------------------------------------
// Kernel 2: flash attention with gated CPB bias.
// Block = 128 threads (4 waves). Wave w owns q rows [qb, qb+16).
// 64-key chunks staged once per block in LDS as f16 (cvt_pkrtz), then two
// 32-key softmax rounds; WMMA operands are raw ds_load_b128; cross-half
// exchanges are v_permlanex16_b32. Staging uses uniform scalar base +
// 32-bit thread offset + immediate offsets (saddr/voffset form).
// ---------------------------------------------------------------------------
__global__ __launch_bounds__(128) void flex_attn_kernel(
    const float* __restrict__ q, const float* __restrict__ kmat,
    const float* __restrict__ v, const float* __restrict__ mu,
    const float* __restrict__ gamma, const float* __restrict__ bt,
    const int* __restrict__ idx_table, float* __restrict__ out) {
  // K chunk, row-major f16:  kth[kk][d],  row stride 72 halves = 144B.
  // V chunk, transposed f16: vth[d][kk],  row stride 72 halves = 144B.
  __shared__ _Float16 kth[64][72];
  __shared__ _Float16 vth[D_][72];
  __shared__ __align__(16) float mks[64];   // sg * mu_k for the chunk

  const int tid  = threadIdx.x;
  const int wave = tid >> 5;
  const int lane = tid & 31;
  const int ln   = lane & 15;   // C/D column (q), A row (k or d)
  const int hi   = lane >> 4;   // half-wave selector

  const int h = blockIdx.y;
  const int b = blockIdx.z;
  const int qb   = (blockIdx.x * 4 + wave) * 16;
  const int qrow = qb + ln;
  const int qld  = qrow < N_ ? qrow : (N_ - 1);  // clamped for tail tiles

  const size_t bh = (size_t)(b * H_ + h);
  const float* qp   = q    + (bh * N_ + (size_t)qld) * D_;
  const float* kp   = kmat + bh * N_ * D_;
  const float* vp   = v    + bh * N_ * D_;
  const float* mup  = mu   + bh * 2 * N_;        // mu_q row; mu_k at +N_
  const int* idxrow = idx_table + (size_t)qld * N_;

  const float sg  = 1.0f / (1.0f + __expf(-gamma[h]));  // sigmoid(gamma[h])
  const float gmq = sg * mup[qld];                      // sg * mu_q
  const float scale = 0.125f;                           // 1/sqrt(64)

  // Per-thread staging byte offsets (chunk-invariant, 32-bit).
  const int kk0   = tid >> 4;          // K row / V k-pair index at it=0
  const int dq0   = (tid & 15) << 2;   // d-quad
  const int koff0 = (kk0 * D_ + dq0) * 4;        // K: byte offset in chunk
  const int voff0 = (kk0 * 2 * D_ + dq0) * 4;    // V: byte offset in chunk

  // Q^T as WMMA B operand; 1/sqrt(d) folded into the f16 conversion.
  v16h bq[2];
#pragma unroll
  for (int j = 0; j < 2; ++j) {
    const float* p0 = qp + 32 * j + 8 * hi;
    f32x4 f0 = *(const f32x4*)(p0 + 0);
    f32x4 f1 = *(const f32x4*)(p0 + 4);
    f32x4 f2 = *(const f32x4*)(p0 + 16);
    f32x4 f3 = *(const f32x4*)(p0 + 20);
    union { v16h v; uint32_t u[8]; } r;
    r.u[0] = pk2(f0.x * scale, f0.y * scale);
    r.u[1] = pk2(f0.z * scale, f0.w * scale);
    r.u[2] = pk2(f1.x * scale, f1.y * scale);
    r.u[3] = pk2(f1.z * scale, f1.w * scale);
    r.u[4] = pk2(f2.x * scale, f2.y * scale);
    r.u[5] = pk2(f2.z * scale, f2.w * scale);
    r.u[6] = pk2(f3.x * scale, f3.y * scale);
    r.u[7] = pk2(f3.z * scale, f3.w * scale);
    bq[j] = r.v;
  }

  // out^T accumulators: acc[t] holds d rows 16t+8*hi+i (i=0..7), col q=qb+ln.
  v8f acc[4] = {};
  float m_run = -3.0e38f;
  float l_run = 0.0f;

  const int nchunk = (N_ + 63) / 64;  // 17 chunks of 64 keys

  for (int c = 0; c < nchunk; ++c) {
    const int kb64 = c * 64;
    // Uniform (scalar) chunk bases: one SALU add per chunk each.
    const char* kpc = (const char*)(kp + (size_t)kb64 * D_);
    const char* vpc = (const char*)(vp + (size_t)kb64 * D_);

    __syncthreads();
    if (kb64 + 64 <= N_) {
      // ---- fast staging: saddr + voffset + immediate ----------------------
#pragma unroll
      for (int it = 0; it < 8; ++it) {          // K: 64 rows x 16 d-quads
        f32x4 kv = *(const f32x4*)(kpc + koff0 + it * (8 * D_ * 4));
        u32x2 w; w.x = pk2(kv.x, kv.y); w.y = pk2(kv.z, kv.w);
        *(u32x2*)&kth[kk0 + 8 * it][dq0] = w;   // one ds_store_b64
      }
#pragma unroll
      for (int it = 0; it < 4; ++it) {          // V: 32 k-pairs x 16 d-quads
        const char* p = vpc + voff0 + it * (16 * D_ * 4);
        f32x4 r0 = *(const f32x4*)(p);
        f32x4 r1 = *(const f32x4*)(p + D_ * 4);
        int kpr = kk0 + 8 * it;
        *(uint32_t*)&vth[dq0 + 0][2 * kpr] = pk2(r0.x, r1.x);
        *(uint32_t*)&vth[dq0 + 1][2 * kpr] = pk2(r0.y, r1.y);
        *(uint32_t*)&vth[dq0 + 2][2 * kpr] = pk2(r0.z, r1.z);
        *(uint32_t*)&vth[dq0 + 3][2 * kpr] = pk2(r0.w, r1.w);
      }
    } else {
      // ---- tail staging: clamped row indices (runs once) ------------------
#pragma unroll
      for (int it = 0; it < 8; ++it) {
        int kk = kk0 + 8 * it;
        int kr = kb64 + kk; if (kr > N_ - 1) kr = N_ - 1;
        f32x4 kv = *(const f32x4*)(kp + (size_t)kr * D_ + dq0);
        u32x2 w; w.x = pk2(kv.x, kv.y); w.y = pk2(kv.z, kv.w);
        *(u32x2*)&kth[kk][dq0] = w;
      }
#pragma unroll
      for (int it = 0; it < 4; ++it) {
        int kpr = kk0 + 8 * it;
        int k0 = kb64 + 2 * kpr;
        int k0c = k0     > N_ - 1 ? N_ - 1 : k0;
        int k1c = k0 + 1 > N_ - 1 ? N_ - 1 : k0 + 1;
        f32x4 r0 = *(const f32x4*)(vp + (size_t)k0c * D_ + dq0);
        f32x4 r1 = *(const f32x4*)(vp + (size_t)k1c * D_ + dq0);
        *(uint32_t*)&vth[dq0 + 0][2 * kpr] = pk2(r0.x, r1.x);
        *(uint32_t*)&vth[dq0 + 1][2 * kpr] = pk2(r0.y, r1.y);
        *(uint32_t*)&vth[dq0 + 2][2 * kpr] = pk2(r0.z, r1.z);
        *(uint32_t*)&vth[dq0 + 3][2 * kpr] = pk2(r0.w, r1.w);
      }
    }
    if (tid < 64) {                             // sg * mu_k, shared by block
      int kr = kb64 + tid; if (kr > N_ - 1) kr = N_ - 1;
      mks[tid] = sg * mup[N_ + kr];
    }
    if (kb64 + 64 < N_) {                       // pull next chunk toward L2
      __builtin_prefetch(vpc + (64 * D_ * 4) + tid * 128, 0, 1);
      __builtin_prefetch(kpc + (64 * D_ * 4) + tid * 128, 0, 1);
    }
    __syncthreads();

    // ---- two 32-key softmax rounds over the staged chunk ------------------
#pragma unroll
    for (int half = 0; half < 2; ++half) {
      const int kb  = kb64 + 32 * half;   // global k base of this round
      const int kro = 32 * half;          // LDS k offset of this round
      const bool full = (kb + 31 <= N_ - 1);   // uniform across the block

      float sv[2][8];
      float tmax = -3.0e38f;
#pragma unroll
      for (int s2 = 0; s2 < 2; ++s2) {
        const _Float16* krl = &kth[kro + s2 * 16 + ln][0];
        v16h a0 = ldsA(krl, 8 * hi);         // d 0..31 slice (A layout)
        v16h a1 = ldsA(krl, 32 + 8 * hi);    // d 32..63 slice
        v8f cc = {};
        cc = wmma_f16(a0, bq[0], cc);
        cc = wmma_f16(a1, bq[1], cc);

        // cc[i] = scaled scores^T[k = kb + s2*16 + 8*hi + i][q = qb+ln]
        const int kcol0 = kb + s2 * 16 + 8 * hi;
        const int kl0   = kro + s2 * 16 + 8 * hi;
        f32x4 g0 = *(const f32x4*)&mks[kl0];       // ds_load_b128
        f32x4 g1 = *(const f32x4*)&mks[kl0 + 4];
        float gv[8] = {g0.x, g0.y, g0.z, g0.w, g1.x, g1.y, g1.z, g1.w};
        if (full) {
#pragma unroll
          for (int i = 0; i < 8; ++i) {
            int id = idxrow[kcol0 + i];                    // unconditional
            float bias = bt[(id < 0 ? 0 : id) * H_ + h];   // unconditional
            float sc = cc[i];
            if (id >= 0) sc = fmaf(gv[i] + gmq, bias, sc); // cndmask
            sv[s2][i] = sc;
            tmax = fmaxf(tmax, sc);
          }
        } else {
#pragma unroll
          for (int i = 0; i < 8; ++i) {
            const int kidx = kcol0 + i;
            float sc = cc[i];
            int id = -1;
            if (kidx < N_) id = idxrow[kidx];
            if (id >= 0) sc = fmaf(gv[i] + gmq, bt[id * H_ + h], sc);
            if (kidx >= N_) sc = -3.0e38f;                 // tail mask
            sv[s2][i] = sc;
            tmax = fmaxf(tmax, sc);
          }
        }
      }

      // ---- online softmax (lane pair L / L+16 share the same q) -----------
      float tm = fmaxf(tmax, xs16f(tmax));     // v_permlanex16_b32
      float nm = fmaxf(m_run, tm);

      float rsum = 0.0f;
      uint32_t pkk[2][4];
#pragma unroll
      for (int s2 = 0; s2 < 2; ++s2) {
#pragma unroll
        for (int j = 0; j < 4; ++j) {
          float p0 = __expf(sv[s2][2 * j + 0] - nm);
          float p1 = __expf(sv[s2][2 * j + 1] - nm);
          rsum += p0 + p1;
          pkk[s2][j] = pk2(p0, p1);
        }
      }
      float osum = rsum + xs16f(rsum);

      // Rescale only when some lane's max actually moved (wave-uniform skip).
      if (__any(nm > m_run)) {
        float corr = __expf(m_run - nm);
        l_run = fmaf(l_run, corr, osum);
#pragma unroll
        for (int t = 0; t < 4; ++t)
#pragma unroll
          for (int i = 0; i < 8; ++i) acc[t][i] *= corr;
      } else {
        l_run += osum;
      }
      m_run = nm;

      // ---- P^T as WMMA B operand (own lane values, no swap) ---------------
      union { v16h v16; uint32_t u[8]; } bp;
#pragma unroll
      for (int j = 0; j < 4; ++j) {
        bp.u[j]     = pkk[0][j];
        bp.u[4 + j] = pkk[1][j];
      }

      // ---- out^T += V^T * P^T (4 d-tiles), operands straight from LDS -----
#pragma unroll
      for (int t = 0; t < 4; ++t) {
        v16h av = ldsA(&vth[16 * t + ln][0], kro + 8 * hi);
        acc[t] = wmma_f16(av, bp.v16, acc[t]);
      }
    }
  }

  // ---- epilogue: out[b,h,q,d] = acc^T / l ---------------------------------
  if (qrow < N_) {
    float inv = 1.0f / l_run;
    float* op = out + (bh * N_ + (size_t)qrow) * D_;
#pragma unroll
    for (int t = 0; t < 4; ++t) {
      f32x4 r0, r1;
      r0.x = acc[t][0] * inv; r0.y = acc[t][1] * inv;
      r0.z = acc[t][2] * inv; r0.w = acc[t][3] * inv;
      r1.x = acc[t][4] * inv; r1.y = acc[t][5] * inv;
      r1.z = acc[t][6] * inv; r1.w = acc[t][7] * inv;
      *(f32x4*)(op + 16 * t + 8 * hi)     = r0;
      *(f32x4*)(op + 16 * t + 8 * hi + 4) = r1;
    }
  }
}

extern "C" void kernel_launch(void* const* d_in, const int* in_sizes, int n_in,
                              void* d_out, int out_size, void* d_ws, size_t ws_size,
                              hipStream_t stream) {
  (void)in_sizes; (void)n_in; (void)out_size; (void)ws_size;
  const float* q     = (const float*)d_in[0];
  const float* k     = (const float*)d_in[1];
  const float* v     = (const float*)d_in[2];
  const float* mu    = (const float*)d_in[3];
  const float* w1    = (const float*)d_in[4];
  const float* b1    = (const float*)d_in[5];
  const float* w2    = (const float*)d_in[6];
  const float* gamma = (const float*)d_in[7];
  const float* rel   = (const float*)d_in[8];
  const int*   idx   = (const int*)d_in[9];
  float* out = (float*)d_out;
  float* bt  = (float*)d_ws;   // 3969*12 floats (~190 KB) of scratch

  cpb_mlp_kernel<<<(TBL_ + 127) / 128, 128, 0, stream>>>(rel, w1, b1, w2, bt);

  dim3 grid((N_ + 63) / 64, H_, B_);   // 17 x 12 x 4 blocks, 4 waves each
  flex_attn_kernel<<<grid, 128, 0, stream>>>(q, k, v, mu, gamma, bt, idx, out);
}
